// BinaryTreeLSTM_15960098472360
// MI455X (gfx1250) — matrix operand order repaired and checked by reading, not compiled, not run on hardware
//
#include <hip/hip_runtime.h>
#include <cstdint>
#include <cstddef>

// ---------------- problem constants (match reference) ----------------
constexpr int TDEPTH  = 10;
constexpr int NNODES  = 1023;   // 2^10 - 1
constexpr int BATCH   = 256;
constexpr int IN_DIM  = 128;
constexpr int HID     = 128;
constexpr int GDIM    = IN_DIM + 2 * HID;   // 384
constexpr int NOUT    = 5 * HID;            // 640 (i, fl, fr, o, u stacked)
constexpr int NCLS    = 5;
constexpr int LDS_STRIDE = GDIM + 8;        // 392 halves -> 784B row, bank-conflict free
constexpr int MT      = 4;                  // M-tiles (of 16 rows) per block
constexpr int MROWS   = MT * 16;            // 64 rows per block

typedef __attribute__((ext_vector_type(16))) _Float16 v16h;
typedef __attribute__((ext_vector_type(8)))  _Float16 h8v;
typedef __attribute__((ext_vector_type(8)))  float    v8f;
typedef __attribute__((ext_vector_type(4)))  float    f4v;

union Frag { v16h v; h8v h[2]; };

__device__ __forceinline__ float sigm(float v) {
    return 1.0f / (1.0f + __expf(-v));
}

// ---------------- weight prep: fp32 -> f16, gate-stacked ----------------
__global__ __launch_bounds__(256) void prep_weights(
    const float* __restrict__ Wi,  const float* __restrict__ Wfl,
    const float* __restrict__ Wfr, const float* __restrict__ Wo,
    const float* __restrict__ Wu,
    const float* __restrict__ bi,  const float* __restrict__ bfl,
    const float* __restrict__ bfr, const float* __restrict__ bo,
    const float* __restrict__ bu,
    _Float16* __restrict__ Wh, float* __restrict__ bias)
{
    int idx = blockIdx.x * blockDim.x + threadIdx.x;
    const float* Ws[5] = {Wi, Wfl, Wfr, Wo, Wu};
    const float* bs[5] = {bi, bfl, bfr, bo, bu};
    if (idx < NOUT * GDIM) {
        int r  = idx / GDIM;          // output column 0..639
        int k  = idx - r * GDIM;
        int g  = r >> 7;              // gate 0..4
        int rr = r & 127;             // hidden unit
        Wh[idx] = (_Float16)Ws[g][rr * GDIM + k];
    }
    if (idx < NOUT) {
        bias[idx] = bs[idx >> 7][idx & 127];
    }
}

// ---------------- one tree level: GEMM (WMMA f16->f32) + LSTM cell ----------------
// Block = 64 rows (64 batches of one node) x 640 cols; 8 waves.
// Wave w owns hidden cols [w*16, w*16+16) across all 5 gates, for 4 M-tiles.
// B-fragments (weights) are loaded once per K-step and reused by all 4 M-tiles.
__global__ __launch_bounds__(256) void tree_level(
    const float*    __restrict__ x,
    const _Float16* __restrict__ Wh,      // [640][384] f16
    const float*    __restrict__ bias,    // [640]
    const _Float16* __restrict__ h_prev,  // [2n][B][H] f16
    const float*    __restrict__ c_prev,  // [2n][B][H] f32
    _Float16*       __restrict__ h_cur,   // [n][B][H] f16
    float*          __restrict__ c_cur,   // [n][B][H] f32
    int start, int leaf)
{
    __shared__ _Float16 smem[MROWS * LDS_STRIDE];   // 64 x 392 halves = 50176 B

    const int t     = threadIdx.x;
    const int node  = blockIdx.x >> 2;          // 4 blocks per node (64 rows each)
    const int bbase = (blockIdx.x & 3) * MROWS; // batch base for this block

    // ---- stage comb tile (64 rows x 384) into LDS as f16 ----
    {
        const int rbase = t >> 4;  // 0..15
        const int c16   = t & 15;  // 0..15 -> 8 halves each per section
#pragma unroll
        for (int rg = 0; rg < MT; ++rg) {
            const int row = rg * 16 + rbase;
            const int b   = bbase + row;
            _Float16* dst = smem + row * LDS_STRIDE;

            // x section [0,128)
            const float* xp = x + ((size_t)b * NNODES + start + node) * IN_DIM + c16 * 8;
            f4v x0 = *(const f4v*)xp;
            f4v x1 = *(const f4v*)(xp + 4);
            h8v hx;
            hx[0] = (_Float16)x0[0]; hx[1] = (_Float16)x0[1];
            hx[2] = (_Float16)x0[2]; hx[3] = (_Float16)x0[3];
            hx[4] = (_Float16)x1[0]; hx[5] = (_Float16)x1[1];
            hx[6] = (_Float16)x1[2]; hx[7] = (_Float16)x1[3];
            *(h8v*)(dst + c16 * 8) = hx;

            if (leaf) {
                h8v z = {};
                *(h8v*)(dst + 128 + c16 * 8) = z;
                *(h8v*)(dst + 256 + c16 * 8) = z;
            } else {
                const size_t lrow = ((size_t)(2 * node)     * BATCH + b) * HID;
                const size_t rrow = ((size_t)(2 * node + 1) * BATCH + b) * HID;
                *(h8v*)(dst + 128 + c16 * 8) = *(const h8v*)(h_prev + lrow + c16 * 8);
                *(h8v*)(dst + 256 + c16 * 8) = *(const h8v*)(h_prev + rrow + c16 * 8);
            }
        }
    }
    __syncthreads();

    const int wv   = t >> 5;        // wave 0..7
    const int lane = t & 31;
    const int l16  = lane & 15;
    const int g2   = lane >> 4;     // lane half-group 0/1
    const int hcol = wv * 16 + l16; // hidden column owned by this lane

    // gfx1250 prefetch of this wave's weight rows (global_prefetch_b8)
    __builtin_prefetch(Wh + (size_t)hcol * GDIM, 0, 1);

    v8f acc[MT][5] = {};

    // ---- K loop: 12 steps of 32; 5 B-frag loads feed MT*5 = 20 WMMAs ----
    const _Float16* arow0 = smem + l16 * LDS_STRIDE;
    for (int kk = 0; kk < GDIM; kk += 32) {
        // B fragments (32x16 f16): lane holds col n = gt*128+hcol, K = {g2*8..+7, 16+g2*8..+7}
        Frag bf[5];
#pragma unroll
        for (int gt = 0; gt < 5; ++gt) {
            const _Float16* wrow = Wh + (size_t)(gt * HID + hcol) * GDIM + kk + g2 * 8;
            bf[gt].h[0] = *(const h8v*)(wrow);
            bf[gt].h[1] = *(const h8v*)(wrow + 16);
        }
#pragma unroll
        for (int tt = 0; tt < MT; ++tt) {
            // A fragment (16x32 f16): lane (m=l16,g=g2) holds K = {g*8..+7, 16+g*8..+7}
            const _Float16* arow = arow0 + tt * 16 * LDS_STRIDE;
            Frag a;
            a.h[0] = *(const h8v*)(arow + kk + g2 * 8);
            a.h[1] = *(const h8v*)(arow + kk + 16 + g2 * 8);
#pragma unroll
            for (int gt = 0; gt < 5; ++gt) {
                acc[tt][gt] = __builtin_amdgcn_wmma_f32_16x16x32_f16(
                    false, a.v, false, bf[gt].v, (short)0, acc[tt][gt], false, false);
            }
        }
    }

    // ---- LSTM cell update, fully in registers (all 5 gates in this wave) ----
    const float bi_  = bias[          hcol];
    const float bfl_ = bias[    HID + hcol];
    const float bfr_ = bias[2 * HID + hcol];
    const float bo_  = bias[3 * HID + hcol];
    const float bu_  = bias[4 * HID + hcol];

#pragma unroll
    for (int tt = 0; tt < MT; ++tt) {
        const int rowb = bbase + tt * 16;
        const size_t outb = ((size_t)node * BATCH + rowb) * HID + hcol;
        const size_t clb  = ((size_t)(2 * node)     * BATCH + rowb) * HID + hcol;
        const size_t crb  = ((size_t)(2 * node + 1) * BATCH + rowb) * HID + hcol;
#pragma unroll
        for (int j = 0; j < 8; ++j) {
            const int m = j + g2 * 8;   // C/D layout: vgpr j, lane-half g2 -> row m
            float iv = sigm(acc[tt][0][j] + bi_);
            float fl = sigm(acc[tt][1][j] + bfl_);
            float fr = sigm(acc[tt][2][j] + bfr_);
            float ov = sigm(acc[tt][3][j] + bo_);
            float uv = tanhf(acc[tt][4][j] + bu_);
            float cl = 0.0f, cr = 0.0f;
            if (!leaf) {
                cl = c_prev[clb + (size_t)m * HID];
                cr = c_prev[crb + (size_t)m * HID];
            }
            float cn = iv * uv + fl * cl + fr * cr;
            float hv = ov * tanhf(cn);
            c_cur[outb + (size_t)m * HID] = cn;
            h_cur[outb + (size_t)m * HID] = (_Float16)hv;
        }
    }
}

// ---------------- root classifier: logits = root_h @ Wcls^T + bcls ----------------
__global__ __launch_bounds__(256) void classify(
    const _Float16* __restrict__ rh,     // [B][H] f16 (node 0 of final level)
    const float* __restrict__ Wc,        // [C][H]
    const float* __restrict__ bc,        // [C]
    float* __restrict__ out)             // [B][C]
{
    int idx = blockIdx.x * blockDim.x + threadIdx.x;
    if (idx >= BATCH * NCLS) return;
    int b  = idx / NCLS;
    int cc = idx - b * NCLS;
    float s = bc[cc];
    const _Float16* hb = rh + (size_t)b * HID;
    const float*    wr = Wc + (size_t)cc * HID;
#pragma unroll 4
    for (int k = 0; k < HID; ++k) s += (float)hb[k] * wr[k];
    out[idx] = s;
}

// ---------------- host-side launcher ----------------
extern "C" void kernel_launch(void* const* d_in, const int* in_sizes, int n_in,
                              void* d_out, int out_size, void* d_ws, size_t ws_size,
                              hipStream_t stream)
{
    (void)in_sizes; (void)n_in; (void)out_size; (void)ws_size;

    const float* x    = (const float*)d_in[0];
    const float* Wi   = (const float*)d_in[1];
    const float* bi   = (const float*)d_in[2];
    const float* Wfl  = (const float*)d_in[3];
    const float* bfl  = (const float*)d_in[4];
    const float* Wfr  = (const float*)d_in[5];
    const float* bfr  = (const float*)d_in[6];
    const float* Wo   = (const float*)d_in[7];
    const float* bo   = (const float*)d_in[8];
    const float* Wu   = (const float*)d_in[9];
    const float* bu   = (const float*)d_in[10];
    const float* Wcls = (const float*)d_in[11];
    const float* bcls = (const float*)d_in[12];

    uint8_t* ws = (uint8_t*)d_ws;
    size_t off = 0;
    _Float16* Wh = (_Float16*)(ws + off); off += (size_t)NOUT * GDIM * sizeof(_Float16);
    float* bias  = (float*)(ws + off);    off += (size_t)NOUT * sizeof(float);
    off = (off + 255) & ~(size_t)255;

    const size_t hElems = (size_t)512 * BATCH * HID;   // max nodes per level = 512
    _Float16* hA = (_Float16*)(ws + off); off += hElems * sizeof(_Float16);
    _Float16* hB = (_Float16*)(ws + off); off += hElems * sizeof(_Float16);
    float*    cA = (float*)(ws + off);    off += hElems * sizeof(float);
    float*    cB = (float*)(ws + off);    off += hElems * sizeof(float);

    prep_weights<<<(NOUT * GDIM + 255) / 256, 256, 0, stream>>>(
        Wi, Wfl, Wfr, Wo, Wu, bi, bfl, bfr, bo, bu, Wh, bias);

    _Float16* hp = hB; _Float16* hc = hA;
    float*    cp = cB; float*    cc = cA;
    for (int d = TDEPTH - 1; d >= 0; --d) {
        int n = 1 << d;
        tree_level<<<4 * n, 256, 0, stream>>>(
            x, Wh, bias, hp, cp, hc, cc, n - 1, (d == TDEPTH - 1) ? 1 : 0);
        _Float16* th = hp; hp = hc; hc = th;
        float*    tc = cp; cp = cc; cc = tc;
    }
    // after the final swap, the d=0 result lives in hp (node 0 only -> [B][H])
    classify<<<(BATCH * NCLS + 255) / 256, 256, 0, stream>>>(
        hp, Wcls, bcls, (float*)d_out);
}
